// AttentionBlock_71356586655956
// MI455X (gfx1250) — matrix operand order repaired and checked by reading, not compiled, not run on hardware
//
#include <hip/hip_runtime.h>
#include <hip/hip_bf16.h>

typedef __attribute__((ext_vector_type(16))) _Float16 v16h;
typedef __attribute__((ext_vector_type(8)))  _Float16 v8h;
typedef __attribute__((ext_vector_type(8)))  float    v8f;
typedef __attribute__((ext_vector_type(4)))  int      v4i;

#define C_DIM 128
#define HW    4096
#define BATCH 4
#define NTHIRD 3
#define INV_SQRT_C 0.08838834764831845f   // 1/sqrt(128)

// LDS tile strides (halfs), padded for bank-conflict-free B-fragment reads
#define KSTR 136    // 68 DW row stride: key*68 % 64 = 4*key -> 16 distinct banks
#define VSTR 40     // 20 DW row stride: c*20 % 64 -> 16 distinct banks

// ---------------- workspace layout (bytes) ----------------
#define WS_ALPHA    0            // 2 * f32
#define WS_MU       256          // 128 * f32
#define WS_RSTD     1024         // 128 * f32
#define WS_BIASEFF  1536         // 384 * f32
#define WS_WQKV     4096         // 3*128*128 f16 = 98304
#define WS_WOUT     102400       // 128*128 f16  = 32768
#define WS_XN       135168       // 4*4096*128 f16 = 4194304
#define WS_Q        4329472
#define WS_K        8523776
#define WS_VT       12718080
#define WS_O        16912384

#if defined(__has_builtin)
#if __has_builtin(__builtin_amdgcn_global_load_async_to_lds_b128)
#define ASYNC_COPY 1
#endif
#endif

static __device__ inline v8f wmma_f16(v16h a, v16h b, v8f c) {
  return __builtin_amdgcn_wmma_f32_16x16x32_f16(false, a, false, b, (short)0, c,
                                                false, false);
}

// A-fragment (16x32 f16, M x K): 16 contiguous halfs (k = 16*half + s), run1 = run0 + 8 halfs
static __device__ inline v16h a_from_runs(const _Float16* p0) {
  v8h lo = *(const v8h*)p0;
  v8h hi = *(const v8h*)(p0 + 8);
  v16h a;
#pragma unroll
  for (int i = 0; i < 8; ++i) { a[i] = lo[i]; a[i + 8] = hi[i]; }
  return a;
}

// B-fragment: 16 contiguous halfs, loaded as two 16B-aligned v8h reads
static __device__ inline v16h b_from_contig(const _Float16* p) {
  v8h lo = *(const v8h*)p;
  v8h hi = *(const v8h*)(p + 8);
  v16h b;
#pragma unroll
  for (int i = 0; i < 8; ++i) { b[i] = lo[i]; b[i + 8] = hi[i]; }
  return b;
}

// 16-byte global -> LDS copy: async (ASYNCcnt) on CDNA5, sync fallback otherwise
static __device__ inline void copy16(const _Float16* g, _Float16* l) {
#ifdef ASYNC_COPY
  __builtin_amdgcn_global_load_async_to_lds_b128(
      (__attribute__((address_space(1))) v4i*)g,
      (__attribute__((address_space(3))) v4i*)l, 0, 0);
#else
  *(uint4*)l = *(const uint4*)g;
#endif
}

static __device__ inline void wait_async0() {
#ifdef ASYNC_COPY
#if __has_builtin(__builtin_amdgcn_s_wait_asynccnt)
  __builtin_amdgcn_s_wait_asynccnt(0);
#else
  asm volatile("s_wait_asynccnt 0x0" ::: "memory");
#endif
#endif
}

// ---------------- K0: alpha = mean(|w|) for both ternary weights ------------
__global__ void k_alpha(const float* __restrict__ wq, const float* __restrict__ wo,
                        float* __restrict__ alphas) {
  __shared__ float red[256];
  const int t = threadIdx.x;
  float s = 0.f;
  for (int i = t; i < NTHIRD * C_DIM * C_DIM; i += 256) s += fabsf(wq[i]);
  red[t] = s; __syncthreads();
  for (int k = 128; k > 0; k >>= 1) { if (t < k) red[t] += red[t + k]; __syncthreads(); }
  if (t == 0) alphas[0] = red[0] / (float)(NTHIRD * C_DIM * C_DIM);
  __syncthreads();
  s = 0.f;
  for (int i = t; i < C_DIM * C_DIM; i += 256) s += fabsf(wo[i]);
  red[t] = s; __syncthreads();
  for (int k = 128; k > 0; k >>= 1) { if (t < k) red[t] += red[t + k]; __syncthreads(); }
  if (t == 0) alphas[1] = red[0] / (float)(C_DIM * C_DIM);
}

// ---------------- K1: W_eff = H^T * tern(W_qkv)  and  tern(W_out) -----------
__global__ void k_weights(const float* __restrict__ wq, const float* __restrict__ wo,
                          const float* __restrict__ alphas,
                          _Float16* __restrict__ weff_qkv,
                          _Float16* __restrict__ weff_out) {
  const int idx = blockIdx.x * 256 + threadIdx.x;
  if (idx < NTHIRD * C_DIM * C_DIM) {
    const int t = idx / (C_DIM * C_DIM);
    const int rem = idx % (C_DIM * C_DIM);
    const int d = rem / C_DIM, c = rem % C_DIM;
    const float alpha = alphas[0], thr = 0.001f * alpha;
    int acc = 0;
#pragma unroll 8
    for (int o = 0; o < C_DIM; ++o) {
      const float w = wq[(t * C_DIM + o) * C_DIM + c];
      int s = (w > thr) - (w < -thr);
      acc += (__popc((unsigned)(o & d)) & 1) ? -s : s;
    }
    weff_qkv[idx] = (_Float16)((float)acc * alpha * INV_SQRT_C);
  } else if (idx < NTHIRD * C_DIM * C_DIM + C_DIM * C_DIM) {
    const int j = idx - NTHIRD * C_DIM * C_DIM;
    const float alpha = alphas[1], thr = 0.001f * alpha;
    const float w = wo[j];
    const int s = (w > thr) - (w < -thr);
    weff_out[j] = (_Float16)(alpha * (float)s);
  }
}

// ---------------- K1b: bias_eff = b_qkv * H (per third) ---------------------
__global__ void k_bias(const float* __restrict__ bq, float* __restrict__ bias_eff) {
  const int idx = blockIdx.x * 256 + threadIdx.x;
  if (idx >= NTHIRD * C_DIM) return;
  const int t = idx / C_DIM, d = idx % C_DIM;
  float s = 0.f;
  for (int o = 0; o < C_DIM; ++o) {
    const float v = bq[t * C_DIM + o];
    s += (__popc((unsigned)(o & d)) & 1) ? -v : v;
  }
  bias_eff[idx] = s * INV_SQRT_C;
}

// ---------------- K2a: GroupNorm statistics ---------------------------------
__global__ void k_gnstats(const float* __restrict__ x, float* __restrict__ mu,
                          float* __restrict__ rstd) {
  __shared__ float rs[256], rq[256];
  const int bg = blockIdx.x;            // b*32 + g
  const int b = bg >> 5, g = bg & 31;
  const float* base = x + ((size_t)b * C_DIM + g * 4) * HW;
  float s = 0.f, q = 0.f;
  for (int i = threadIdx.x; i < 4 * HW; i += 256) {
    const float v = base[(i >> 12) * HW + (i & (HW - 1))];
    s += v; q += v * v;
  }
  rs[threadIdx.x] = s; rq[threadIdx.x] = q; __syncthreads();
  for (int k = 128; k > 0; k >>= 1) {
    if (threadIdx.x < k) { rs[threadIdx.x] += rs[threadIdx.x + k];
                           rq[threadIdx.x] += rq[threadIdx.x + k]; }
    __syncthreads();
  }
  if (threadIdx.x == 0) {
    const float m = rs[0] / (float)(4 * HW);
    const float var = rq[0] / (float)(4 * HW) - m * m;
    mu[bg] = m;
    rstd[bg] = rsqrtf(var + 1e-5f);
  }
}

// ---------------- K2b: normalize + transpose to xn[b,hw,c] f16 --------------
__global__ void k_gnapply(const float* __restrict__ x, const float* __restrict__ gamma,
                          const float* __restrict__ beta, const float* __restrict__ mu,
                          const float* __restrict__ rstd, _Float16* __restrict__ xn) {
  const int idx = blockIdx.x * 256 + threadIdx.x;   // ((b*HW+p)*C + c)
  const int c = idx & (C_DIM - 1);
  const int p = (idx >> 7) & (HW - 1);
  const int b = idx >> 19;
  const int bg = b * 32 + (c >> 2);
  const float v = x[((size_t)b * C_DIM + c) * HW + p];
  xn[idx] = (_Float16)((v - mu[bg]) * rstd[bg] * gamma[c] + beta[c]);
}

// ---------------- K3: QKV = xn @ W_eff^T (WMMA), write q,k,v^T --------------
__global__ void __launch_bounds__(256) k_qkv(const _Float16* __restrict__ xn,
                                             const _Float16* __restrict__ weff,
                                             const float* __restrict__ bias_eff,
                                             _Float16* __restrict__ qb,
                                             _Float16* __restrict__ kb,
                                             _Float16* __restrict__ vT) {
  const int wave = threadIdx.x >> 5, lane = threadIdx.x & 31;
  const int m = lane & 15, h = lane >> 4;
  const int gp0 = blockIdx.x * 128 + wave * 16;     // global pixel (b*HW+p)
  const int bIdx = gp0 >> 12;

  v16h af[4];
#pragma unroll
  for (int s = 0; s < 4; ++s)
    af[s] = a_from_runs(xn + (size_t)(gp0 + m) * C_DIM + s * 32 + h * 16);

#pragma unroll
  for (int ot = 0; ot < 24; ++ot) {
    const int t = ot >> 3;
    const int o_in = (ot & 7) * 16 + m;             // output channel (N = lane%16)
    v8f acc = {};
#pragma unroll
    for (int s = 0; s < 4; ++s) {
      const v16h bf = *(const v16h*)(weff + ((size_t)t * C_DIM + o_in) * C_DIM
                                          + s * 32 + h * 16);
      acc = wmma_f16(af[s], bf, acc);
    }
    const float bias = bias_eff[t * C_DIM + o_in];
#pragma unroll
    for (int r = 0; r < 8; ++r) {
      const int pix = gp0 + r + h * 8;              // C/D: M = r + 8*half
      const float val = acc[r] + bias;
      if (t == 0)      qb[(size_t)pix * C_DIM + o_in] = (_Float16)val;
      else if (t == 1) kb[(size_t)pix * C_DIM + o_in] = (_Float16)val;
      else             vT[((size_t)bIdx * C_DIM + o_in) * HW + (pix & (HW - 1))] =
                           (_Float16)val;
    }
  }
}

// ---------------- K4: flash attention, async LDS double-buffered ------------
__global__ void __launch_bounds__(256) k_attn(const _Float16* __restrict__ qbuf,
                                              const _Float16* __restrict__ kbuf,
                                              const _Float16* __restrict__ vT,
                                              _Float16* __restrict__ obuf) {
  __shared__ __align__(16) _Float16 Kl[2][32][KSTR];    // keys x channels
  __shared__ __align__(16) _Float16 Vl[2][C_DIM][VSTR]; // channels x keys
  __shared__ __align__(16) _Float16 plds[8][16][32];

  const int tid = threadIdx.x;
  const int wave = tid >> 5, lane = tid & 31;
  const int m = lane & 15, h = lane >> 4;
  const int bIdx = blockIdx.y;
  const int q0 = blockIdx.x * 128 + wave * 16;
  const _Float16* kbase = kbuf + (size_t)bIdx * HW * C_DIM;
  const _Float16* vbase = vT + (size_t)bIdx * C_DIM * HW;

  // copy-assignment for this thread (same every key block)
  const int kkey = tid >> 3, kseg = tid & 7;   // K: 32 keys x 8 x 32B segments
  const int vc = tid >> 1, vseg = tid & 1;     // V: 128 rows x 2 x 32B segments

  v16h qf[4];
#pragma unroll
  for (int s = 0; s < 4; ++s)
    qf[s] = a_from_runs(qbuf + ((size_t)bIdx * HW + q0 + m) * C_DIM + s * 32 + h * 16);

  v8f accO[8];
#pragma unroll
  for (int t = 0; t < 8; ++t) accO[t] = (v8f){};
  float mrun[8], lrun[8];
#pragma unroll
  for (int r = 0; r < 8; ++r) { mrun[r] = -1e30f; lrun[r] = 0.f; }

  const int NKB = HW / 32;

  // prologue: stage block 0 into slot 0
  {
    const _Float16* ksrc = kbase + (size_t)kkey * C_DIM + kseg * 16;
    copy16(ksrc,     &Kl[0][kkey][kseg * 16]);
    copy16(ksrc + 8, &Kl[0][kkey][kseg * 16 + 8]);
    const _Float16* vsrc = vbase + (size_t)vc * HW + vseg * 16;
    copy16(vsrc,     &Vl[0][vc][vseg * 16]);
    copy16(vsrc + 8, &Vl[0][vc][vseg * 16 + 8]);
  }
  wait_async0();
  __syncthreads();

  for (int kb = 0; kb < NKB; ++kb) {
    const int slot = kb & 1;
    const int k0 = kb * 32;

    // stage next block into the other slot (overlaps with compute below)
    if (kb + 1 < NKB) {
      const int nslot = slot ^ 1;
      const int nk0 = k0 + 32;
      const _Float16* ksrc = kbase + (size_t)(nk0 + kkey) * C_DIM + kseg * 16;
      copy16(ksrc,     &Kl[nslot][kkey][kseg * 16]);
      copy16(ksrc + 8, &Kl[nslot][kkey][kseg * 16 + 8]);
      const _Float16* vsrc = vbase + (size_t)vc * HW + nk0 + vseg * 16;
      copy16(vsrc,     &Vl[nslot][vc][vseg * 16]);
      copy16(vsrc + 8, &Vl[nslot][vc][vseg * 16 + 8]);
    }

    // ---- S = Q @ K^T (two 16-key N-tiles) from LDS ----
    v8f s0 = {}, s1 = {};
#pragma unroll
    for (int s = 0; s < 4; ++s) {
      const v16h b0 = b_from_contig(&Kl[slot][m][s * 32 + h * 16]);
      const v16h b1 = b_from_contig(&Kl[slot][16 + m][s * 32 + h * 16]);
      s0 = wmma_f16(qf[s], b0, s0);
      s1 = wmma_f16(qf[s], b1, s1);
    }

    // ---- online softmax (rows live in half-wave lanes) ----
#pragma unroll
    for (int r = 0; r < 8; ++r) {
      float a0 = s0[r] * INV_SQRT_C, a1 = s1[r] * INV_SQRT_C;
      float mx = fmaxf(a0, a1);
      mx = fmaxf(mx, __shfl_xor(mx, 1, 32));
      mx = fmaxf(mx, __shfl_xor(mx, 2, 32));
      mx = fmaxf(mx, __shfl_xor(mx, 4, 32));
      mx = fmaxf(mx, __shfl_xor(mx, 8, 32));
      const float mnew = fmaxf(mrun[r], mx);
      const float corr = __expf(mrun[r] - mnew);
      const float p0 = __expf(a0 - mnew), p1 = __expf(a1 - mnew);
      float rsum = p0 + p1;
      rsum += __shfl_xor(rsum, 1, 32);
      rsum += __shfl_xor(rsum, 2, 32);
      rsum += __shfl_xor(rsum, 4, 32);
      rsum += __shfl_xor(rsum, 8, 32);
      lrun[r] = lrun[r] * corr + rsum;
      mrun[r] = mnew;
#pragma unroll
      for (int t = 0; t < 8; ++t) accO[t][r] *= corr;
      const int row = r + h * 8;
      plds[wave][row][m]      = (_Float16)p0;
      plds[wave][row][m + 16] = (_Float16)p1;
    }
    asm volatile("s_wait_dscnt 0x0" ::: "memory");   // DS RAW within wave

    // ---- re-read P as A-fragment, O += P @ V from LDS ----
    const v16h pf = a_from_runs(&plds[wave][m][h * 16]);
#pragma unroll
    for (int t = 0; t < 8; ++t) {
      const v16h bv = b_from_contig(&Vl[slot][t * 16 + m][h * 16]);
      accO[t] = wmma_f16(pf, bv, accO[t]);
    }

    // next block's copies must be complete (per-wave) before the barrier
    wait_async0();
    __syncthreads();
  }

  // ---- finalize: divide by l, store O[b,hw,c] f16 ----
#pragma unroll
  for (int r = 0; r < 8; ++r) {
    const float inv = 1.f / lrun[r];
    const size_t rowbase = ((size_t)bIdx * HW + q0 + r + h * 8) * C_DIM;
#pragma unroll
    for (int t = 0; t < 8; ++t)
      obuf[rowbase + t * 16 + m] = (_Float16)(accO[t][r] * inv);
  }
}

// ---------------- K5: out = O @ tern(w_out)^T + b_out + x -------------------
__global__ void __launch_bounds__(256) k_proj(const _Float16* __restrict__ obuf,
                                              const _Float16* __restrict__ wout,
                                              const float* __restrict__ bout,
                                              const float* __restrict__ xin,
                                              float* __restrict__ out) {
  const int wave = threadIdx.x >> 5, lane = threadIdx.x & 31;
  const int m = lane & 15, h = lane >> 4;
  const int gp0 = blockIdx.x * 128 + wave * 16;
  const int bIdx = gp0 >> 12, ploc = gp0 & (HW - 1);

  v16h of[4];
#pragma unroll
  for (int s = 0; s < 4; ++s)
    of[s] = a_from_runs(obuf + (size_t)(gp0 + m) * C_DIM + s * 32 + h * 16);

#pragma unroll
  for (int ot = 0; ot < 8; ++ot) {
    const int o = ot * 16 + m;
    v8f acc = {};
#pragma unroll
    for (int s = 0; s < 4; ++s) {
      const v16h bf = *(const v16h*)(wout + (size_t)o * C_DIM + s * 32 + h * 16);
      acc = wmma_f16(of[s], bf, acc);
    }
    const float bias = bout[o];
#pragma unroll
    for (int r = 0; r < 8; ++r) {
      const size_t addr = ((size_t)bIdx * C_DIM + o) * HW + ploc + r + h * 8;
      out[addr] = acc[r] + bias + xin[addr];
    }
  }
}

// ---------------- launcher ---------------------------------------------------
extern "C" void kernel_launch(void* const* d_in, const int* in_sizes, int n_in,
                              void* d_out, int out_size, void* d_ws, size_t ws_size,
                              hipStream_t stream) {
  (void)in_sizes; (void)n_in; (void)out_size; (void)ws_size;
  const float* x     = (const float*)d_in[0];
  const float* gamma = (const float*)d_in[1];
  const float* beta  = (const float*)d_in[2];
  const float* w_qkv = (const float*)d_in[3];
  const float* b_qkv = (const float*)d_in[4];
  const float* w_out = (const float*)d_in[5];
  const float* b_out = (const float*)d_in[6];
  float* out = (float*)d_out;

  char* ws = (char*)d_ws;
  float*    alphas   = (float*)(ws + WS_ALPHA);
  float*    mu       = (float*)(ws + WS_MU);
  float*    rstd     = (float*)(ws + WS_RSTD);
  float*    bias_eff = (float*)(ws + WS_BIASEFF);
  _Float16* weff_qkv = (_Float16*)(ws + WS_WQKV);
  _Float16* weff_out = (_Float16*)(ws + WS_WOUT);
  _Float16* xn       = (_Float16*)(ws + WS_XN);
  _Float16* qb       = (_Float16*)(ws + WS_Q);
  _Float16* kb       = (_Float16*)(ws + WS_K);
  _Float16* vT       = (_Float16*)(ws + WS_VT);
  _Float16* obuf     = (_Float16*)(ws + WS_O);

  k_alpha<<<1, 256, 0, stream>>>(w_qkv, w_out, alphas);
  k_weights<<<(NTHIRD * C_DIM * C_DIM + C_DIM * C_DIM) / 256, 256, 0, stream>>>(
      w_qkv, w_out, alphas, weff_qkv, weff_out);
  k_bias<<<2, 256, 0, stream>>>(b_qkv, bias_eff);
  k_gnstats<<<BATCH * 32, 256, 0, stream>>>(x, mu, rstd);
  k_gnapply<<<(BATCH * HW * C_DIM) / 256, 256, 0, stream>>>(x, gamma, beta, mu, rstd, xn);
  k_qkv<<<(BATCH * HW) / 128, 256, 0, stream>>>(xn, weff_qkv, bias_eff, qb, kb, vT);
  k_attn<<<dim3(HW / 128, BATCH), 256, 0, stream>>>(qb, kb, vT, obuf);
  k_proj<<<(BATCH * HW) / 128, 256, 0, stream>>>(obuf, weff_out, b_out, x, out);
}
